// PointTransformerDecoder_21912923144351
// MI455X (gfx1250) — matrix operand order repaired and checked by reading, not compiled, not run on hardware
//
#include <hip/hip_runtime.h>
#include <hip/hip_bf16.h>
#include <stdint.h>

// ---------------------------------------------------------------------------
// PointTransformerDecoder on MI455X (gfx1250, wave32, WMMA bf16)
//   B=256, FEAT_IN=512, HID=64, SEQ=128, OUT=768 (=256 points * 3 dims)
// ---------------------------------------------------------------------------

typedef __attribute__((ext_vector_type(16))) __bf16 v16bf;
typedef __attribute__((ext_vector_type(8)))  float  v8f;

#define WMMA_BF16(c, a, b) \
  (c) = __builtin_amdgcn_wmma_f32_16x16x32_bf16(false, (a), false, (b), (short)0, (c), false, false)

__device__ __forceinline__ __bf16 bf(float x) { return (__bf16)x; }

// Async global->LDS copy, 16 bytes per lane, tracked on ASYNCcnt.
// Flat LDS addresses map to LDS by truncation (ISA 10.2), so the low 32 bits
// of a generic __shared__ pointer are the LDS byte address operand.
__device__ __forceinline__ void async_g2lds_b128(void* lds_ptr, const void* gptr) {
  const uint32_t lds = (uint32_t)(uintptr_t)lds_ptr;
  const unsigned long long ga = (unsigned long long)(uintptr_t)gptr;
  asm volatile("global_load_async_to_lds_b128 %0, %1, off"
               :: "v"(lds), "v"(ga) : "memory");
}
__device__ __forceinline__ void wait_asynccnt0() {
  asm volatile("s_wait_asynccnt 0x0" ::: "memory");
}

// --- wave32 WMMA fragment loaders (layouts per CDNA5 ISA 7.12.2) ------------

// A: 16x32 bf16 tile from row-major storage (row stride = lda elements).
// lanes 0-15 hold M=lane, K pairs {0..7,16..23}; lanes 16-31 hold K {8..15,24..31}.
__device__ __forceinline__ v16bf load_a_bf16(const __bf16* A, int lda, int lane) {
  const int m  = lane & 15;
  const int hi = lane >> 4;
  v16bf a;
#pragma unroll
  for (int p = 0; p < 8; ++p) {
    const int k = ((p < 4) ? (2 * p) : (2 * p + 8)) + hi * 8;
    a[2 * p]     = A[m * lda + k];
    a[2 * p + 1] = A[m * lda + k + 1];
  }
  return a;
}

// Same pattern but gathering f32 and converting to bf16 on the fly.
__device__ __forceinline__ v16bf load_a_f32(const float* A, int lda, int lane) {
  const int m  = lane & 15;
  const int hi = lane >> 4;
  v16bf a;
#pragma unroll
  for (int p = 0; p < 8; ++p) {
    const int k = ((p < 4) ? (2 * p) : (2 * p + 8)) + hi * 8;
    a[2 * p]     = bf(A[m * lda + k]);
    a[2 * p + 1] = bf(A[m * lda + k + 1]);
  }
  return a;
}

// B: 32x16 tile from *n-major* ("transposed") storage BT[n*ldb + k].
// Each lane reads k = {0..7,16..23} + 8*hi (matches A fragment order).
__device__ __forceinline__ v16bf load_bt_bf16(const __bf16* BT, int ldb, int lane) {
  const int n  = lane & 15;
  const int hi = lane >> 4;
  v16bf b;
#pragma unroll
  for (int p = 0; p < 8; ++p) {
    const int k = ((p < 4) ? (2 * p) : (2 * p + 8)) + hi * 8;
    b[2 * p]     = BT[n * ldb + k];
    b[2 * p + 1] = BT[n * ldb + k + 1];
  }
  return b;
}

// B: 32x16 tile straight from row-major f32 global B[k*ldb + n].
__device__ __forceinline__ v16bf load_b_f32rm(const float* B, int ldb, int lane) {
  const int n  = lane & 15;
  const int hi = lane >> 4;
  v16bf b;
#pragma unroll
  for (int p = 0; p < 8; ++p) {
    const int k = ((p < 4) ? (2 * p) : (2 * p + 8)) + hi * 8;
    b[2 * p]     = bf(B[k * ldb + n]);
    b[2 * p + 1] = bf(B[(k + 1) * ldb + n]);
  }
  return b;
}

// 16x16 tile of [16,64] @ [64,16]  (K=64 -> two WMMAs)
__device__ __forceinline__ v8f gemm_k64(const __bf16* A, int lda, const __bf16* BT, int lane) {
  v8f c = {};
  v16bf a0 = load_a_bf16(A, lda, lane);
  v16bf b0 = load_bt_bf16(BT, 64, lane);
  WMMA_BF16(c, a0, b0);
  v16bf a1 = load_a_bf16(A + 32, lda, lane);
  v16bf b1 = load_bt_bf16(BT + 32, 64, lane);
  WMMA_BF16(c, a1, b1);
  return c;
}

// ---------------------------------------------------------------------------
// Kernel A: h0 = relu(x @ W0 + b0), stored bf16.  M=256, K=512, N=64.
// 16 blocks (one 16-row M tile) x 4 waves (one 16-col N tile each).
// ---------------------------------------------------------------------------
__global__ __launch_bounds__(128)
void h0_kernel(const float* __restrict__ x, const float* __restrict__ W0,
               const float* __restrict__ b0, __bf16* __restrict__ h0) {
  const int lane = threadIdx.x & 31;
  const int wave = threadIdx.x >> 5;
  const int m0 = blockIdx.x * 16;
  const int n0 = wave * 16;
  v8f c = {};
  for (int kk = 0; kk < 512; kk += 32) {
    v16bf a = load_a_f32(x + m0 * 512 + kk, 512, lane);
    v16bf b = load_b_f32rm(W0 + kk * 64 + n0, 64, lane);
    WMMA_BF16(c, a, b);
  }
  const int hi = lane >> 4;
  const int col = n0 + (lane & 15);
#pragma unroll
  for (int r = 0; r < 8; ++r) {
    const int row = m0 + r + hi * 8;
    float v = c[r] + b0[col];
    h0[row * 64 + col] = bf(v > 0.f ? v : 0.f);
  }
}

// ---------------------------------------------------------------------------
// Kernel B: 128-step autoregressive scan. One WG per 16 batch rows; the whole
// cache slice (16 x 129 x 64 bf16 = 258 KB) stays resident in WGP LDS.
// Algebra: scores = cache . (Wk @ q) + bk.q ;  attn.v = (sum_s attn*cache) @ Wv + bv
// so per step we run 4 small WMMA GEMMs + two O(S*H) VALU passes.
// ---------------------------------------------------------------------------
__global__ __launch_bounds__(256)
void decode_kernel(const __bf16* __restrict__ h0,
                   const float* __restrict__ Wq, const float* __restrict__ bq,
                   const float* __restrict__ Wk, const float* __restrict__ bk,
                   const float* __restrict__ Wv, const float* __restrict__ bv,
                   const float* __restrict__ Wo, const float* __restrict__ bo,
                   __bf16* __restrict__ xseq) {
  extern __shared__ char smem[];
  __bf16* cacheL = (__bf16*)smem;            // 16*129*64
  __bf16* WqT  = cacheL + 16 * 129 * 64;     // 64x64, n-major (= Wq transposed)
  __bf16* WkT  = WqT + 4096;                 // 64x64, row-major Wk (== (Wk^T) n-major)
  __bf16* WvT  = WkT + 4096;
  __bf16* WoT  = WvT + 4096;
  __bf16* qb   = WoT + 4096;                 // 16x64 bf16
  __bf16* ctxb = qb + 1024;                  // 16x64 bf16
  __bf16* hb   = ctxb + 1024;                // 16x64 bf16
  float*  qk   = (float*)(hb + 1024);        // 16x64
  float*  sc   = qk + 1024;                  // 16x132 (padded)
  float*  sb   = sc + 16 * 132;              // 16

  const int tid  = threadIdx.x;
  const int lane = tid & 31;
  const int wave = tid >> 5;
  const int b0g  = blockIdx.x * 16;

  // cache[:,0] = h0
  for (int i = tid; i < 16 * 64; i += 256) {
    const int b = i >> 6, h = i & 63;
    cacheL[(b * 129) * 64 + h] = h0[(b0g + b) * 64 + h];
  }
  __syncthreads();

  for (int t = 0; t < 128; ++t) {
    // Prefetch next step's weights into L2 while this step computes.
    if (tid == 0 && t + 1 < 128) {
      __builtin_prefetch(Wq + (t + 1) * 4096, 0, 1);
      __builtin_prefetch(Wk + (t + 1) * 4096, 0, 1);
      __builtin_prefetch(Wv + (t + 1) * 4096, 0, 1);
      __builtin_prefetch(Wo + (t + 1) * 4096, 0, 1);
    }
    // Stage this step's weights into LDS as bf16 in B-operand (n-major) layout.
    for (int i = tid; i < 4096; i += 256) {
      const int n = i >> 6, k = i & 63;
      WqT[i] = bf(Wq[t * 4096 + k * 64 + n]);
      WkT[i] = bf(Wk[t * 4096 + i]);          // Wk row-major == (Wk^T) n-major
      WvT[i] = bf(Wv[t * 4096 + k * 64 + n]);
      WoT[i] = bf(Wo[t * 4096 + k * 64 + n]);
    }
    __syncthreads();

    // q = cache[:,t] @ Wq + bq
    if (wave < 4) {
      const int n0 = wave * 16;
      v8f c = gemm_k64(cacheL + t * 64, 129 * 64, WqT + n0 * 64, lane);
      const int hi = lane >> 4, col = n0 + (lane & 15);
#pragma unroll
      for (int r = 0; r < 8; ++r)
        qb[(r + hi * 8) * 64 + col] = bf(c[r] + bq[t * 64 + col]);
    }
    __syncthreads();

    // qk = q @ Wk^T   (and sb[b] = bk . q[b] in parallel on wave 4)
    if (wave < 4) {
      const int n0 = wave * 16;
      v8f c = gemm_k64(qb, 64, WkT + n0 * 64, lane);
      const int hi = lane >> 4, col = n0 + (lane & 15);
#pragma unroll
      for (int r = 0; r < 8; ++r)
        qk[(r + hi * 8) * 64 + col] = c[r];
    } else if (wave == 4 && lane < 16) {
      float acc = 0.f;
      for (int h = 0; h < 64; ++h) acc += (float)qb[lane * 64 + h] * bk[t * 64 + h];
      sb[lane] = acc;
    }
    __syncthreads();

    // scores[b,s] = (cache[b,s] . qk[b] + sb[b]) / sqrt(64),  s = 0..t
    const int tp1 = t + 1;
    for (int i = tid; i < 16 * tp1; i += 256) {
      const int b = i / tp1, s = i % tp1;
      const __bf16* row = cacheL + (b * 129 + s) * 64;
      float acc = 0.f;
      for (int h = 0; h < 64; ++h) acc += (float)row[h] * qk[b * 64 + h];
      sc[b * 132 + s] = (acc + sb[b]) * 0.125f;
    }
    __syncthreads();

    // softmax over s (rows are short; one thread per batch row)
    if (tid < 16) {
      float mx = -3.4e38f;
      for (int s = 0; s < tp1; ++s) mx = fmaxf(mx, sc[tid * 132 + s]);
      float sum = 0.f;
      for (int s = 0; s < tp1; ++s) {
        const float e = __expf(sc[tid * 132 + s] - mx);
        sc[tid * 132 + s] = e;
        sum += e;
      }
      const float inv = 1.f / sum;
      for (int s = 0; s < tp1; ++s) sc[tid * 132 + s] *= inv;
    }
    __syncthreads();

    // ctx[b] = sum_s attn[b,s] * cache[b,s]
    for (int i = tid; i < 1024; i += 256) {
      const int b = i >> 6, h = i & 63;
      float acc = 0.f;
      for (int s = 0; s < tp1; ++s)
        acc += sc[b * 132 + s] * (float)cacheL[(b * 129 + s) * 64 + h];
      ctxb[i] = bf(acc);
    }
    __syncthreads();

    // h = relu(ctx @ Wv + bv)
    if (wave < 4) {
      const int n0 = wave * 16;
      v8f c = gemm_k64(ctxb, 64, WvT + n0 * 64, lane);
      const int hi = lane >> 4, col = n0 + (lane & 15);
#pragma unroll
      for (int r = 0; r < 8; ++r) {
        const float v = c[r] + bv[t * 64 + col];
        hb[(r + hi * 8) * 64 + col] = bf(v > 0.f ? v : 0.f);
      }
    }
    __syncthreads();

    // new = cache[:,0] + h @ Wo + bo ; append to cache and emit to x_seq
    if (wave < 4) {
      const int n0 = wave * 16;
      v8f c = gemm_k64(hb, 64, WoT + n0 * 64, lane);
      const int hi = lane >> 4, col = n0 + (lane & 15);
#pragma unroll
      for (int r = 0; r < 8; ++r) {
        const int row = r + hi * 8;
        const float nv = (float)cacheL[(row * 129) * 64 + col] + c[r] + bo[t * 64 + col];
        const __bf16 nb = bf(nv);
        cacheL[(row * 129 + t + 1) * 64 + col] = nb;
        xseq[((b0g + row) * 128 + t) * 64 + col] = nb;
      }
    }
    __syncthreads();
  }
}

// ---------------------------------------------------------------------------
// Kernel C: y = x_seq @ Wf + bf.  M = 256*128 = 32768, K = 64, N = 768.
// Grid (2048 M-tiles, 12 N-chunks of 64). The Wf chunk (64x64 f32, 256-B
// contiguous rows) is pulled into LDS with GLOBAL_LOAD_ASYNC_TO_LDS_B128
// (ASYNCcnt-tracked, no VGPR round-trip), then converted to bf16 n-major.
// Pure HBM-write-bound (100 MB f32 out) -> stores go straight to global.
// ---------------------------------------------------------------------------
__global__ __launch_bounds__(256)
void proj_kernel(const __bf16* __restrict__ xseq, const float* __restrict__ Wf,
                 const float* __restrict__ bfv, float* __restrict__ y) {
  __shared__ float  Wst[64 * 64];   // raw f32 staging (k-major), async-filled
  __shared__ __bf16 WfT[64 * 64];   // bf16 n-major B operand
  const int tid = threadIdx.x, lane = tid & 31, wave = tid >> 5;
  const int m0  = blockIdx.x * 16;
  const int nch = blockIdx.y * 64;

  // Async-stage Wf[k=0..63, nch..nch+63]: 4 x B128 (4 f32) per thread.
#pragma unroll
  for (int j = 0; j < 4; ++j) {
    const int f = (j * 256 + tid) * 4;      // flat f32 index, 4-aligned in-row
    const int k = f >> 6, n = f & 63;
    async_g2lds_b128(&Wst[f], Wf + k * 768 + nch + n);
  }
  wait_asynccnt0();
  __syncthreads();

  // Convert to bf16, transposing to n-major for the B-fragment loader.
  for (int i = tid; i < 4096; i += 256) {
    const int n = i >> 6, k = i & 63;
    WfT[i] = bf(Wst[k * 64 + n]);
  }
  __syncthreads();

  if (wave < 4) {
    const int n0 = wave * 16;
    v8f c = gemm_k64(xseq + m0 * 64, 64, WfT + n0 * 64, lane);
    const int hi = lane >> 4, col = nch + n0 + (lane & 15);
#pragma unroll
    for (int r = 0; r < 8; ++r)
      y[(m0 + r + hi * 8) * 768 + col] = c[r] + bfv[col];
  }
}

// ---------------------------------------------------------------------------
extern "C" void kernel_launch(void* const* d_in, const int* in_sizes, int n_in,
                              void* d_out, int out_size, void* d_ws, size_t ws_size,
                              hipStream_t stream) {
  const float* x   = (const float*)d_in[0];
  const float* W0  = (const float*)d_in[1];
  const float* b0  = (const float*)d_in[2];
  const float* Wq  = (const float*)d_in[3];
  const float* bq  = (const float*)d_in[4];
  const float* Wk  = (const float*)d_in[5];
  const float* bk  = (const float*)d_in[6];
  const float* Wv  = (const float*)d_in[7];
  const float* bv  = (const float*)d_in[8];
  const float* Wo  = (const float*)d_in[9];
  const float* bo  = (const float*)d_in[10];
  const float* Wf  = (const float*)d_in[11];
  const float* bfv = (const float*)d_in[12];
  float* y = (float*)d_out;

  __bf16* h0b  = (__bf16*)d_ws;                              // 256*64 bf16
  __bf16* xseq = (__bf16*)((char*)d_ws + 256 * 64 * 2);      // 256*128*64 bf16

  h0_kernel<<<16, 128, 0, stream>>>(x, W0, b0, h0b);

  const size_t smem = (size_t)(16 * 129 * 64 + 4 * 4096 + 3 * 1024) * sizeof(__bf16)
                    + (size_t)(1024 + 16 * 132 + 16) * sizeof(float);   // 315,712 B
  decode_kernel<<<16, 256, smem, stream>>>(h0b, Wq, bq, Wk, bk, Wv, bv, Wo, bo, xseq);

  proj_kernel<<<dim3(2048, 12), 256, 0, stream>>>(xseq, Wf, bfv, y);
}
